// VRNN_31327491457280
// MI455X (gfx1250) — matrix operand order repaired and checked by reading, not compiled, not run on hardware
//
#include <hip/hip_runtime.h>
#include <stdint.h>
#include <stddef.h>

// ---------------- types ----------------
typedef __bf16 bf16_t;
typedef __attribute__((ext_vector_type(16))) __bf16 v16bf;
typedef __attribute__((ext_vector_type(8)))  __bf16 v8bf;
typedef __attribute__((ext_vector_type(8)))  float  v8f;
typedef __attribute__((ext_vector_type(4)))  unsigned int u32x4;
typedef __attribute__((ext_vector_type(8)))  int i32x8;
typedef __attribute__((ext_vector_type(4)))  int i32x4;

// ---------------- problem dims ----------------
constexpr int BB = 256;   // batch
constexpr int TT = 256;   // time steps
constexpr int DD = 512;   // data dim
constexpr int HH = 1024;  // hidden
constexpr int ZZ = 128;   // latent
constexpr float LEPS = 1e-6f;

// ---------------- helpers ----------------
__device__ __forceinline__ bf16_t f32_to_bf16(float f) {
    unsigned u = __builtin_bit_cast(unsigned, f);
    unsigned r = u + 0x7FFFu + ((u >> 16) & 1u);   // round-to-nearest-even
    unsigned short h = (unsigned short)(r >> 16);
    return __builtin_bit_cast(bf16_t, h);
}

__device__ __forceinline__ float fast_sigmoid(float v) {
    return __builtin_amdgcn_rcpf(1.0f + __expf(-v));          // v_rcp_f32 + v_exp_f32
}

__device__ __forceinline__ float fast_tanh(float v) {
    float x = fminf(fmaxf(v, -15.0f), 15.0f);
    float e2 = __expf(2.0f * x);
    return (e2 - 1.0f) * __builtin_amdgcn_rcpf(e2 + 1.0f);
}

__device__ __forceinline__ float act_apply(float v, int act) {
    if (act == 1) return fmaxf(v, 0.0f);                                   // relu
    if (act == 2) { float a = fabsf(v);                                    // softplus
                    return log1pf(__expf(-a)) + fmaxf(v, 0.0f); }
    if (act == 3) return fast_sigmoid(v);                                  // sigmoid
    return v;
}

// ---------------- Tensor Data Mover tile load ----------------
#if __has_include(<hip/amd_detail/amd_gfx1250_TDM.h>)
#define TDM_SIX_ARGS 1
#endif

constexpr int BM = 128, BN = 128, KB = 32;
constexpr int KSTEP = 64;  // K per pipeline stage (2 sub-slabs of 32)
constexpr int LDT = 40;    // LDS row stride (bf16 elems) = KB + 8 pad
constexpr int NSTG = 3;    // TDM pipeline stages

// 2D tile (tile_dim0=32 elems contiguous, tile_dim1=rows) from a K-major
// matrix, DMA'd into LDS with hardware padding: 16 DWORDs (one 32-elem bf16
// row) then 4 DWORDs pad -> LDS row stride of 40 bf16 elems.
__device__ __forceinline__ void tdm_load_tile(const bf16_t* gsrc, bf16_t* lds_dst,
                                              int rows, int stride_elems) {
    unsigned long long ga = (unsigned long long)(uintptr_t)gsrc;
    unsigned int la = (unsigned int)(uintptr_t)lds_dst;   // LDS aperture: low 32 bits
    u32x4 g0;
    g0.x = 1u;                                            // count=1 valid user descriptor
    g0.y = la;                                            // lds_addr [63:32]
    g0.z = (unsigned int)(ga & 0xFFFFFFFFu);              // global_addr [95:64]
    g0.w = (unsigned int)((ga >> 32) & 0x01FFFFFFu)       // global_addr [120:96]
         | (2u << 30);                                    // type=2 (image)
    unsigned int td0 = (unsigned int)stride_elems;        // tensor_dim0 (OOB bound)
    unsigned int td1 = 0x100000u;                         // tensor_dim1 (ample)
    unsigned long long s0 = (unsigned long long)stride_elems;
    i32x8 g1;
    g1[0] = (int)((1u << 16)        // data_size = 2 bytes
                | (1u << 20)        // pad_enable
                | (3u << 22)        // pad_interval: 16 DWORDs
                | (3u << 25));      // pad_amount: 4 DWORDs
    g1[1] = (int)((td0 & 0xFFFFu) << 16);                             // dim0[15:0]
    g1[2] = (int)(((td0 >> 16) & 0xFFFFu) | ((td1 & 0xFFFFu) << 16)); // dim0[31:16], dim1[15:0]
    g1[3] = (int)(((td1 >> 16) & 0xFFFFu) | ((unsigned)KB << 16));    // dim1[31:16], tile_dim0
    g1[4] = (int)(rows & 0xFFFF);                                     // tile_dim1 (tile_dim2=0)
    g1[5] = (int)(s0 & 0xFFFFFFFFu);                                  // dim0_stride[31:0]
    g1[6] = (int)((s0 >> 32) & 0xFFFFu);                              // dim0_stride[47:32]
    g1[7] = 0;
    i32x4 gz = {};
#if defined(TDM_SIX_ARGS)
    i32x8 gz8 = {};
    __builtin_amdgcn_tensor_load_to_lds(g0, g1, gz, gz, gz8, 0);
#else
    __builtin_amdgcn_tensor_load_to_lds(g0, g1, gz, gz, 0);
#endif
}

// ---------------- WMMA GEMM with split-K A sourcing ----------------
// C[256,N] = act( [A1 | A2][256,K] @ Wt[N,K]^T + bias )
// A columns [0,K1) come from A1 (row stride lda1), [K1,K) from A2 (lda2).
// K1 multiple of 32, K multiple of 64. Wt: weights pre-transposed [N][K] bf16.
// 3-stage TDM->LDS pipeline, 64 K-elems per stage (4 tensor ops), DMA up to 2
// stages ahead (s_wait_tensorcnt<=4, executed by ALL waves -- no-op for the
// non-DMA waves, keeps the branch scalar). One barrier per 16 WMMAs per wave.
__global__ __launch_bounds__(256) void gemm_bias_act(
    const bf16_t* __restrict__ A1, int lda1, int K1,
    const bf16_t* __restrict__ A2, int lda2,
    const bf16_t* __restrict__ Wt,
    const float* __restrict__ bias,
    float* __restrict__ Cf, bf16_t* __restrict__ Cb,
    int N, int K, int act)
{
    __shared__ __align__(16) bf16_t As[NSTG][2][BM * LDT];
    __shared__ __align__(16) bf16_t Bs[NSTG][2][BN * LDT];

    const int tid  = threadIdx.x;
    const int lane = tid & 31;
    const int wave = tid >> 5;
    const int m0 = blockIdx.y * BM;
    const int n0 = blockIdx.x * BN;
    const int wm = (wave & 3) * 32;     // 2 m-tiles per wave
    const int wn = (wave >> 2) * 64;    // 4 n-tiles per wave
    const bool tdm_wave = (wave == 0);

    v8f acc[2][4] = {};

    const bf16_t* Bbase = Wt + (size_t)n0 * K;
    const int nk = K / KSTEP;

    auto slab_src = [&](int kk) -> const bf16_t* {
        return (kk < K1) ? (A1 + (size_t)m0 * lda1 + kk)
                         : (A2 + (size_t)m0 * lda2 + (kk - K1));
    };
    auto slab_lda = [&](int kk) -> int { return (kk < K1) ? lda1 : lda2; };

    if (tdm_wave) {                      // prime the pipeline: stages 0..NSTG-2
        #pragma unroll
        for (int s = 0; s < NSTG - 1; ++s) {
            if (s < nk) {
                const int kk = s * KSTEP;
                #pragma unroll
                for (int sub = 0; sub < 2; ++sub) {
                    const int ks = kk + sub * KB;
                    tdm_load_tile(slab_src(ks), &As[s][sub][0], BM, slab_lda(ks));
                    tdm_load_tile(Bbase + ks,   &Bs[s][sub][0], BN, K);
                }
            }
        }
    }

    const int am0 = wm + (lane & 15);
    const int ak  = (lane < 16) ? 0 : 8;
    const int bn0 = wn + (lane & 15);
    const int bk  = (lane < 16) ? 0 : 8;

    for (int it = 0; it < nk; ++it) {
        const int cur = it % NSTG;
        // stage `it` complete; up to one full stage (4 tensor ops) may still fly
        if (it + NSTG - 1 <= nk) __builtin_amdgcn_s_wait_tensorcnt(4 * (NSTG - 2));
        else                     __builtin_amdgcn_s_wait_tensorcnt(0);
        __syncthreads();                 // (a) tile `it` visible (b) buffer reuse safe
        if (tdm_wave && (it + NSTG - 1 < nk)) {
            const int kk = (it + NSTG - 1) * KSTEP;
            const int nxt = (it + NSTG - 1) % NSTG;
            #pragma unroll
            for (int sub = 0; sub < 2; ++sub) {
                const int ks = kk + sub * KB;
                tdm_load_tile(slab_src(ks), &As[nxt][sub][0], BM, slab_lda(ks));
                tdm_load_tile(Bbase + ks,   &Bs[nxt][sub][0], BN, K);
            }
        }

        #pragma unroll
        for (int ks = 0; ks < 2; ++ks) {
            // issue all fragment loads, then the WMMA burst
            v16bf afr[2];
            #pragma unroll
            for (int mt = 0; mt < 2; ++mt) {
                const int am = am0 + mt * 16;
                v8bf lo = *reinterpret_cast<const v8bf*>(&As[cur][ks][am * LDT + ak]);
                v8bf hi = *reinterpret_cast<const v8bf*>(&As[cur][ks][am * LDT + ak + 16]);
                afr[mt] = __builtin_shufflevector(lo, hi,
                    0,1,2,3,4,5,6,7,8,9,10,11,12,13,14,15);
            }
            v16bf bfr[4];
            #pragma unroll
            for (int nt = 0; nt < 4; ++nt) {
                const int bn = bn0 + nt * 16;
                v8bf lo = *reinterpret_cast<const v8bf*>(&Bs[cur][ks][bn * LDT + bk]);
                v8bf hi = *reinterpret_cast<const v8bf*>(&Bs[cur][ks][bn * LDT + bk + 16]);
                bfr[nt] = __builtin_shufflevector(lo, hi,
                    0,1,2,3,4,5,6,7,8,9,10,11,12,13,14,15);
            }
            #pragma unroll
            for (int mt = 0; mt < 2; ++mt)
                #pragma unroll
                for (int nt = 0; nt < 4; ++nt)
                    acc[mt][nt] = __builtin_amdgcn_wmma_f32_16x16x32_bf16(
                        false, afr[mt], false, bfr[nt], (short)0, acc[mt][nt],
                        false, false);
        }
    }

    // fused bias + activation epilogue
    #pragma unroll
    for (int mt = 0; mt < 2; ++mt) {
        #pragma unroll
        for (int nt = 0; nt < 4; ++nt) {
            const int col = n0 + wn + nt * 16 + (lane & 15);
            const float bv = bias[col];
            #pragma unroll
            for (int r = 0; r < 8; ++r) {
                const int row = m0 + wm + mt * 16 + ((lane < 16) ? r : r + 8);
                float v = act_apply(acc[mt][nt][r] + bv, act);
                if (Cf) Cf[(size_t)row * N + col] = v;
                if (Cb) Cb[(size_t)row * N + col] = f32_to_bf16(v);
            }
        }
    }
}

// ---------------- elementwise kernels ----------------
__global__ void f2bf_kernel(const float* __restrict__ in, bf16_t* __restrict__ out, int n) {
    int i = blockIdx.x * 256 + threadIdx.x;
    if (i < n) out[i] = f32_to_bf16(in[i]);
}

// w: [K][N] f32  ->  wt: [N][K] bf16
__global__ void f2bf_t_kernel(const float* __restrict__ w, bf16_t* __restrict__ wt,
                              int K, int N) {
    int i = blockIdx.x * 256 + threadIdx.x;
    if (i >= K * N) return;
    int n = i / K, k = i % K;
    wt[i] = f32_to_bf16(w[(size_t)k * N + n]);
}

__global__ void bias_concat_kernel(const float* __restrict__ a, int na,
                                   const float* __restrict__ b, int nb,
                                   float* __restrict__ out) {
    int i = blockIdx.x * 256 + threadIdx.x;
    if (i < na) out[i] = a[i];
    else if (i < na + nb) out[i] = b[i - na];
}

__global__ void reparam_kernel(const float* __restrict__ zm, const float* __restrict__ zs,
                               const float* __restrict__ eps_t,
                               float* __restrict__ z_out, bf16_t* __restrict__ z_bf) {
    int i = blockIdx.x * 256 + threadIdx.x;
    if (i >= BB * ZZ) return;
    float z = zm[i] + zs[i] * eps_t[i];
    z_out[i] = z;
    z_bf[i]  = f32_to_bf16(z);
}

// GRU with h_prev = 0: h = (1-u)*n
__global__ void gru_kernel(const float* __restrict__ gi, const float* __restrict__ bhh,
                           bf16_t* __restrict__ h_bf) {
    int i = blockIdx.x * 256 + threadIdx.x;
    if (i >= BB * HH) return;
    int b = i / HH, j = i % HH;
    const float* g = gi + (size_t)b * (3 * HH);
    float r = fast_sigmoid(g[j]      + bhh[j]);
    float u = fast_sigmoid(g[HH + j] + bhh[HH + j]);
    float n = fast_tanh(g[2 * HH + j] + r * bhh[2 * HH + j]);
    h_bf[i] = f32_to_bf16((1.0f - u) * n);
}

__global__ __launch_bounds__(256) void loss_rows_kernel(
    const float* __restrict__ pm, const float* __restrict__ ps,
    const float* __restrict__ zm, const float* __restrict__ zs,
    const float* __restrict__ xm, const float* __restrict__ x, int t,
    float* __restrict__ kld_row, float* __restrict__ nll_row)
{
    __shared__ float s1[256];
    __shared__ float s2[256];
    const int b = blockIdx.x, tid = threadIdx.x;
    float kacc = 0.0f;
    for (int z = tid; z < ZZ; z += 256) {
        float psv = ps[b * ZZ + z], zsv = zs[b * ZZ + z];
        float dm  = zm[b * ZZ + z] - pm[b * ZZ + z];
        kacc += logf(psv + LEPS) - logf(zsv + LEPS)
              + (zsv * zsv + dm * dm) * __builtin_amdgcn_rcpf(2.0f * psv * psv + LEPS)
              - 0.5f;
    }
    float nacc = 0.0f;
    for (int d = tid; d < DD; d += 256) {
        float xmv = xm[b * DD + d];
        xmv = fminf(fmaxf(xmv, 1e-6f), 1.0f - 1e-6f);
        float xv = x[((size_t)b * TT + t) * DD + d];
        nacc += xv * logf(xmv) + (1.0f - xv) * log1pf(-xmv);
    }
    s1[tid] = kacc; s2[tid] = nacc;
    __syncthreads();
    for (int off = 128; off > 0; off >>= 1) {
        if (tid < off) { s1[tid] += s1[tid + off]; s2[tid] += s2[tid + off]; }
        __syncthreads();
    }
    if (tid == 0) { kld_row[b] = s1[0]; nll_row[b] = -s2[0]; }
}

__global__ __launch_bounds__(256) void loss_reduce_kernel(
    const float* __restrict__ kld_row, const float* __restrict__ nll_row,
    float* __restrict__ out, int is_last)
{
    __shared__ float s1[256];
    __shared__ float s2[256];
    const int tid = threadIdx.x;       // BB == 256 == blockDim
    s1[tid] = kld_row[tid]; s2[tid] = nll_row[tid];
    __syncthreads();
    for (int off = 128; off > 0; off >>= 1) {
        if (tid < off) { s1[tid] += s1[tid + off]; s2[tid] += s2[tid + off]; }
        __syncthreads();
    }
    if (tid == 0) {
        float kld = s1[0] / (float)BB, nll = s2[0] / (float)BB;
        out[0] += kld + nll;
        if (is_last) { out[1] = kld; out[2] = nll; }
    }
}

__global__ void init_out_kernel(float* out) { if (threadIdx.x == 0) out[0] = 0.0f; }

// ---------------- host driver ----------------
extern "C" void kernel_launch(void* const* d_in, const int* in_sizes, int n_in,
                              void* d_out, int out_size, void* d_ws, size_t ws_size,
                              hipStream_t stream) {
    (void)in_sizes; (void)n_in; (void)out_size; (void)ws_size;
    const float* x      = (const float*)d_in[0];
    const float* eps    = (const float*)d_in[1];
    const float* h0     = (const float*)d_in[2];
    const float* phix_w = (const float*)d_in[3];  const float* phix_b = (const float*)d_in[4];
    const float* pm_w1  = (const float*)d_in[5];  const float* pm_b1  = (const float*)d_in[6];
    const float* pm_w2  = (const float*)d_in[7];  const float* pm_b2  = (const float*)d_in[8];
    const float* ps_w1  = (const float*)d_in[9];  const float* ps_b1  = (const float*)d_in[10];
    const float* ps_w2  = (const float*)d_in[11]; const float* ps_b2  = (const float*)d_in[12];
    const float* em_w1  = (const float*)d_in[13]; const float* em_b1  = (const float*)d_in[14];
    const float* em_w2  = (const float*)d_in[15]; const float* em_b2  = (const float*)d_in[16];
    const float* es_w1  = (const float*)d_in[17]; const float* es_b1  = (const float*)d_in[18];
    const float* es_w2  = (const float*)d_in[19]; const float* es_b2  = (const float*)d_in[20];
    const float* dm_w1  = (const float*)d_in[21]; const float* dm_b1  = (const float*)d_in[22];
    const float* dm_w2  = (const float*)d_in[23]; const float* dm_b2  = (const float*)d_in[24];
    const float* gru_wih = (const float*)d_in[25];
    const float* gru_bih = (const float*)d_in[26];
    const float* gru_bhh = (const float*)d_in[27];

    float* out = (float*)d_out;
    const size_t TBZ = (size_t)TT * BB * ZZ;
    float* Zs_base = out + 3;
    float* Em_base = out + 3 + TBZ;
    float* Es_base = out + 3 + 2 * TBZ;
    float* Dm_base = out + 3 + 3 * TBZ;

    uintptr_t cur = (uintptr_t)d_ws;
    auto alloc = [&](size_t bytes) -> void* {
        void* p = (void*)cur;
        cur += (bytes + 255) & ~(size_t)255;
        return p;
    };
    auto abf = [&](size_t elems) { return (bf16_t*)alloc(elems * sizeof(bf16_t)); };
    auto af  = [&](size_t elems) { return (float*)alloc(elems * sizeof(float)); };

    // bf16 weights, pre-transposed to [N][K]; N-fused pairs share one buffer
    bf16_t* wb_phix  = abf((size_t)DD * HH);
    bf16_t* wb_pmps1 = abf((size_t)2 * HH * HH);        // [pm_w1^T ; ps_w1^T]
    bf16_t* wb_pm2   = abf((size_t)HH * ZZ);
    bf16_t* wb_ps2   = abf((size_t)HH * ZZ);
    bf16_t* wb_emes1 = abf((size_t)2 * (2 * HH) * HH);  // [em_w1^T ; es_w1^T]
    bf16_t* wb_em2   = abf((size_t)HH * ZZ);
    bf16_t* wb_es2   = abf((size_t)HH * ZZ);
    bf16_t* wb_dm1   = abf((size_t)(HH + ZZ) * HH);
    bf16_t* wb_dm2   = abf((size_t)HH * DD);
    bf16_t* wb_gih   = abf((size_t)(HH + ZZ) * 3 * HH);

    // fused biases
    float* b_pmps1 = af((size_t)2 * HH);
    float* b_emes1 = af((size_t)2 * HH);

    // bf16 activations (x converted whole, sliced in place by the GEMM)
    bf16_t* x_bf     = abf((size_t)BB * TT * DD);
    bf16_t* h_bf     = abf((size_t)BB * HH);
    bf16_t* phi_bf   = abf((size_t)BB * HH);
    bf16_t* pmps1_bf = abf((size_t)BB * 2 * HH);        // [pm1 | ps1]
    bf16_t* emes1_bf = abf((size_t)BB * 2 * HH);        // [em1 | es1]
    bf16_t* dm1_bf   = abf((size_t)BB * HH);
    bf16_t* z_bf     = abf((size_t)BB * ZZ);

    // f32 scratch
    float* pm_f    = af((size_t)BB * ZZ);
    float* ps_f    = af((size_t)BB * ZZ);
    float* gi_f    = af((size_t)BB * 3 * HH);
    float* kld_row = af(BB);
    float* nll_row = af(BB);

    auto convT = [&](const float* w, bf16_t* wt, int K, int N) {
        size_t n = (size_t)K * N;
        f2bf_t_kernel<<<dim3((unsigned)((n + 255) / 256)), 256, 0, stream>>>(w, wt, K, N);
    };
    auto gemm2 = [&](const bf16_t* A1, int lda1, int K1, const bf16_t* A2, int lda2,
                     const bf16_t* Wt, const float* bias, float* Cf, bf16_t* Cb,
                     int N, int K, int act) {
        dim3 grid((unsigned)(N / BN), (unsigned)(BB / BM));
        gemm_bias_act<<<grid, 256, 0, stream>>>(A1, lda1, K1, A2, lda2,
                                                Wt, bias, Cf, Cb, N, K, act);
    };
    auto gemm = [&](const bf16_t* A, int lda, const bf16_t* Wt, const float* bias,
                    float* Cf, bf16_t* Cb, int N, int K, int act) {
        gemm2(A, lda, K, A, lda, Wt, bias, Cf, Cb, N, K, act);
    };

    // one-time prep per launch (weights stay resident across the whole scan)
    convT(phix_w, wb_phix, DD, HH);
    convT(pm_w1,  wb_pmps1,                     HH, HH);
    convT(ps_w1,  wb_pmps1 + (size_t)HH * HH,   HH, HH);
    convT(pm_w2,  wb_pm2,  HH, ZZ);
    convT(ps_w2,  wb_ps2,  HH, ZZ);
    convT(em_w1,  wb_emes1,                         2 * HH, HH);
    convT(es_w1,  wb_emes1 + (size_t)HH * 2 * HH,   2 * HH, HH);
    convT(em_w2,  wb_em2,  HH, ZZ);
    convT(es_w2,  wb_es2,  HH, ZZ);
    convT(dm_w1,  wb_dm1,  HH + ZZ, HH);
    convT(dm_w2,  wb_dm2,  HH, DD);
    convT(gru_wih, wb_gih, HH + ZZ, 3 * HH);
    bias_concat_kernel<<<dim3((unsigned)((2 * HH + 255) / 256)), 256, 0, stream>>>(
        pm_b1, HH, ps_b1, HH, b_pmps1);
    bias_concat_kernel<<<dim3((unsigned)((2 * HH + 255) / 256)), 256, 0, stream>>>(
        em_b1, HH, es_b1, HH, b_emes1);
    f2bf_kernel<<<dim3((unsigned)((BB * HH + 255) / 256)), 256, 0, stream>>>(
        h0, h_bf, BB * HH);
    f2bf_kernel<<<dim3((unsigned)(((size_t)BB * TT * DD + 255) / 256)), 256, 0, stream>>>(
        x, x_bf, BB * TT * DD);
    init_out_kernel<<<1, 32, 0, stream>>>(out);

    const unsigned ew_bz = (unsigned)((BB * ZZ + 255) / 256);
    const unsigned ew_bh = (unsigned)((BB * HH + 255) / 256);

    for (int t = 0; t < TT; ++t) {
        float* zm_o = Em_base + (size_t)t * BB * ZZ;
        float* zs_o = Es_base + (size_t)t * BB * ZZ;
        float* z_o  = Zs_base + (size_t)t * BB * ZZ;
        float* xm_o = Dm_base + (size_t)t * BB * DD;
        const float* eps_t = eps + (size_t)t * BB * ZZ;

        // phi = relu(x[:,t,:] @ phix_w + b)  -- x sliced in place via lda = T*D
        gemm(x_bf + (size_t)t * DD, TT * DD, wb_phix, phix_b,
             nullptr, phi_bf, HH, DD, 1);
        // fused prior hidden: [pm1 | ps1] = relu(h @ [pm_w1|ps_w1] + b)
        gemm(h_bf, HH, wb_pmps1, b_pmps1, nullptr, pmps1_bf, 2 * HH, HH, 1);
        gemm(pmps1_bf,      2 * HH, wb_pm2, pm_b2, pm_f, nullptr, ZZ, HH, 1);
        gemm(pmps1_bf + HH, 2 * HH, wb_ps2, ps_b2, ps_f, nullptr, ZZ, HH, 2);
        // fused encoder hidden on virtual concat [h | phi]
        gemm2(h_bf, HH, HH, phi_bf, HH, wb_emes1, b_emes1,
              nullptr, emes1_bf, 2 * HH, 2 * HH, 1);
        gemm(emes1_bf,      2 * HH, wb_em2, em_b2, zm_o, nullptr, ZZ, HH, 1);
        gemm(emes1_bf + HH, 2 * HH, wb_es2, es_b2, zs_o, nullptr, ZZ, HH, 2);
        reparam_kernel<<<ew_bz, 256, 0, stream>>>(zm_o, zs_o, eps_t, z_o, z_bf);
        // decoder on virtual concat [h | z]
        gemm2(h_bf, HH, HH, z_bf, ZZ, wb_dm1, dm_b1, nullptr, dm1_bf, HH, HH + ZZ, 1);
        gemm(dm1_bf, HH, wb_dm2, dm_b2, xm_o, nullptr, DD, HH, 3);
        // GRU input GEMM on virtual concat [phi | z]
        gemm2(phi_bf, HH, HH, z_bf, ZZ, wb_gih, gru_bih, gi_f, nullptr,
              3 * HH, HH + ZZ, 0);
        loss_rows_kernel<<<BB, 256, 0, stream>>>(pm_f, ps_f, zm_o, zs_o, xm_o, x, t,
                                                 kld_row, nll_row);
        loss_reduce_kernel<<<1, 256, 0, stream>>>(kld_row, nll_row, out, t == TT - 1);
        gru_kernel<<<ew_bh, 256, 0, stream>>>(gi_f, gru_bhh, h_bf);
    }
}
